// SuperVLAD_56083682951188
// MI455X (gfx1250) — compile-verified, matrix operand order checked by reading, not run on hardware
//
#include <hip/hip_runtime.h>

typedef __attribute__((ext_vector_type(16))) __bf16 v16bf;
typedef __attribute__((ext_vector_type(8)))  float  v8f;

#define N_IMG  64
#define DIM    128
#define PPIX   4096
#define KOUT   64
#define PT     128
#define SPLITS 8
#define PPB    (PPIX / SPLITS)   // 512 pixels per block

union Frag {
  v16bf v;
  unsigned int u[8];
  unsigned short h[16];
};

__device__ __forceinline__ unsigned short f2bf(float f) {
  unsigned int u = __float_as_uint(f);
  u = u + 0x7FFFu + ((u >> 16) & 1u);   // round-to-nearest-even
  return (unsigned short)(u >> 16);
}
__device__ __forceinline__ float bf2f(unsigned short h) {
  return __uint_as_float(((unsigned int)h) << 16);
}
// K-dim pair base for bf16 A/B fragments (ISA 7.12.2): VGPR v holds K pairs,
// lanes 16-31 use K offset +8 in each 16-wide K half.
__device__ __forceinline__ int kpair(int v, int hf) {
  return ((v < 4) ? (v * 2) : (16 + (v - 4) * 2)) + hf * 8;
}
__device__ __forceinline__ v8f wmma_bf16(v16bf a, v16bf b, v8f c) {
  return __builtin_amdgcn_wmma_f32_16x16x32_bf16(false, a, false, b,
                                                 (short)0, c, false, false);
}

// Kernel 1: per (image, pixel-split) partial VLAD accumulation.
__global__ __launch_bounds__(256)
void supervlad_partial(const float* __restrict__ x,
                       const float* __restrict__ w,
                       float* __restrict__ ws) {
  __shared__ __align__(16) unsigned short sX [PT * DIM];  // [pixel][d] bf16, 32KB
  __shared__ __align__(16) unsigned short sXT[DIM * PT];  // [d][pixel] bf16, 32KB
  __shared__ __align__(16) unsigned short sW [KOUT * DIM];// [k][d]     bf16, 16KB
  __shared__ __align__(16) unsigned short sA [KOUT * PT]; // [k][pixel] bf16, 16KB

  const int tid  = threadIdx.x;
  const int lane = tid & 31;
  const int wv   = tid >> 5;
  const int hf   = lane >> 4;   // lane half (K offset select / C-row select)
  const int l16  = lane & 15;
  const int img  = blockIdx.x >> 3;      // blockIdx / SPLITS
  const int spl  = blockIdx.x & (SPLITS - 1);

  // Preload W rows 0..63 (ghost row 64 handled separately via scalar loads)
  for (int i = tid; i < KOUT * DIM; i += 256) sW[i] = f2bf(w[i]);

  const float* gw = w + KOUT * DIM;                // ghost cluster row
  const float* xb = x + (size_t)img * DIM * PPIX;

  const v8f vz = {0.f, 0.f, 0.f, 0.f, 0.f, 0.f, 0.f, 0.f};
  v8f vacc[4];                                     // persistent vlad tiles
  vacc[0] = vz; vacc[1] = vz; vacc[2] = vz; vacc[3] = vz;

  const int mt    = wv >> 1;        // this wave's vlad cluster M-tile (0..3)
  const int nbase = (wv & 1) * 4;   // this wave's vlad d N-tile base
  const int pg    = wv * 16;        // this wave's 16-pixel group for logits

  const int pstart = spl * PPB;
  const int pend   = pstart + PPB;

  __syncthreads();

  for (int pt0 = pstart; pt0 < pend; pt0 += PT) {
    // ---- phase 1: load + per-pixel L2 normalize; 256 threads, half-D each ----
    {
      const int p  = tid & 127;
      const int d0 = (tid >> 7) * 64;
      float ss = 0.f;
      #pragma unroll 4
      for (int d = d0; d < d0 + 64; ++d) {
        float v = xb[(size_t)d * PPIX + pt0 + p];
        ss += v * v;
        sX[p * DIM + d] = f2bf(v);
      }
      float* spart = (float*)sA;     // sA region free here (consumed last iter)
      spart[tid] = ss;
      __syncthreads();
      float tot = spart[p] + spart[p + 128];
      float sc = 1.f / fmaxf(sqrtf(tot), 1e-12f);
      #pragma unroll 4
      for (int d = d0; d < d0 + 64; ++d) {
        unsigned short b = f2bf(bf2f(sX[p * DIM + d]) * sc);
        sX[p * DIM + d]  = b;        // [p][d] copy: logits B + ghost dot
        sXT[d * PT + p]  = b;        // [d][p] copy: vlad B
      }
    }
    // prefetch next pixel tile while we compute on this one (global_prefetch_b8)
    if (pt0 + PT < pend) {
      const float* nx = xb + pt0 + PT;
      for (int i = tid; i < DIM * 4; i += 256)
        __builtin_prefetch(&nx[(size_t)(i >> 2) * PPIX + (i & 3) * 32], 0, 1);
    }
    __syncthreads();

    // ---- phase 2: logits for this wave's 16 pixels, 4 cluster tiles ----
    v8f lacc[4]; lacc[0] = vz; lacc[1] = vz; lacc[2] = vz; lacc[3] = vz;
    const int px = pg + l16;        // this lane's pixel column (C-layout N)
    #pragma unroll
    for (int s = 0; s < 4; ++s) {   // K-dim (d) steps of 32
      Frag B;
      #pragma unroll
      for (int v = 0; v < 8; ++v) {
        int d0 = s * 32 + kpair(v, hf);
        B.u[v] = *(const unsigned int*)&sX[px * DIM + d0];
      }
      #pragma unroll
      for (int t = 0; t < 4; ++t) {
        Frag A;
        int row = t * 16 + l16;
        #pragma unroll
        for (int v = 0; v < 8; ++v) {
          int d0 = s * 32 + kpair(v, hf);
          A.u[v] = *(const unsigned int*)&sW[row * DIM + d0];
        }
        lacc[t] = wmma_bf16(A.v, B.v, lacc[t]);
      }
    }

    // ghost cluster logit (k=64): uniform W row -> scalar loads
    float g = 0.f;
    #pragma unroll 4
    for (int d = 0; d < DIM; ++d) g += gw[d] * bf2f(sX[px * DIM + d]);

    // ---- softmax over 65 clusters; column split across lanes n and n+16 ----
    float mx = g;
    #pragma unroll
    for (int t = 0; t < 4; ++t)
      #pragma unroll
      for (int r = 0; r < 8; ++r) mx = fmaxf(mx, lacc[t][r]);
    mx = fmaxf(mx, __shfl_xor(mx, 16, 32));
    float den = (hf == 0) ? __expf(g - mx) : 0.f;   // count ghost once
    #pragma unroll
    for (int t = 0; t < 4; ++t)
      #pragma unroll
      for (int r = 0; r < 8; ++r) {
        float e = __expf(lacc[t][r] - mx);
        lacc[t][r] = e;
        den += e;
      }
    den += __shfl_xor(den, 16, 32);
    float inv = 1.f / den;
    #pragma unroll
    for (int t = 0; t < 4; ++t)
      #pragma unroll
      for (int r = 0; r < 8; ++r) {
        int k = t * 16 + hf * 8 + r;               // C-layout row -> cluster
        sA[k * PT + px] = f2bf(lacc[t][r] * inv);
      }
    __syncthreads();

    // ---- phase 3: vlad += A(64xPT) x X^T(PTx128), tiles owned per wave ----
    #pragma unroll
    for (int s = 0; s < 4; ++s) {   // K-dim (pixel) steps of 32
      Frag A;
      int row = mt * 16 + l16;
      #pragma unroll
      for (int v = 0; v < 8; ++v) {
        int p0 = s * 32 + kpair(v, hf);
        A.u[v] = *(const unsigned int*)&sA[row * PT + p0];
      }
      #pragma unroll
      for (int j = 0; j < 4; ++j) {
        Frag B;
        int dcol = (nbase + j) * 16 + l16;
        #pragma unroll
        for (int v = 0; v < 8; ++v) {
          int p0 = s * 32 + kpair(v, hf);
          B.u[v] = *(const unsigned int*)&sXT[dcol * PT + p0];  // contiguous
        }
        vacc[j] = wmma_bf16(A.v, B.v, vacc[j]);
      }
    }
    __syncthreads();
  }

  // ---- store partial vlad (deterministic, no atomics) ----
  float* wsb = ws + ((size_t)(img * SPLITS + spl)) * KOUT * DIM;
  #pragma unroll
  for (int j = 0; j < 4; ++j)
    #pragma unroll
    for (int r = 0; r < 8; ++r) {
      int k = mt * 16 + hf * 8 + r;
      int d = (nbase + j) * 16 + l16;
      wsb[k * DIM + d] = vacc[j][r];
    }
}

// Kernel 2: sum the SPLITS partials, intra-norm rows, global norm, store.
__global__ __launch_bounds__(256)
void supervlad_finalize(const float* __restrict__ ws, float* __restrict__ out) {
  __shared__ __align__(16) float sV[KOUT * DIM];   // 32KB
  __shared__ float sred[2 * KOUT + 1];

  const int img = blockIdx.x;
  const int tid = threadIdx.x;
  const float* base = ws + (size_t)img * SPLITS * KOUT * DIM;

  for (int e = tid; e < KOUT * DIM; e += 256) {
    float s = 0.f;
    #pragma unroll
    for (int sp = 0; sp < SPLITS; ++sp) s += base[(size_t)sp * KOUT * DIM + e];
    sV[e] = s;
  }
  __syncthreads();

  if (tid < KOUT) {
    float ss = 0.f;
    for (int d = 0; d < DIM; ++d) { float v = sV[tid * DIM + d]; ss += v * v; }
    float sc1 = 1.f / fmaxf(sqrtf(ss), 1e-12f);
    sred[tid] = sc1;
    sred[KOUT + tid] = ss * sc1 * sc1;
  }
  __syncthreads();
  if (tid == 0) {
    float tot = 0.f;
    for (int k = 0; k < KOUT; ++k) tot += sred[KOUT + k];
    sred[2 * KOUT] = 1.f / fmaxf(sqrtf(tot), 1e-12f);
  }
  __syncthreads();
  float sc2 = sred[2 * KOUT];
  float* ob = out + (size_t)img * KOUT * DIM;
  for (int e = tid; e < KOUT * DIM; e += 256)
    ob[e] = sV[e] * sred[e >> 7] * sc2;
}

extern "C" void kernel_launch(void* const* d_in, const int* in_sizes, int n_in,
                              void* d_out, int out_size, void* d_ws, size_t ws_size,
                              hipStream_t stream) {
  (void)in_sizes; (void)n_in; (void)out_size; (void)ws_size;
  const float* x = (const float*)d_in[0];   // (64,128,64,64) f32
  const float* w = (const float*)d_in[1];   // (65,128) f32
  float* out = (float*)d_out;               // (64, 64*128) f32
  float* ws  = (float*)d_ws;                // needs 64*8*8192*4 = 16 MB scratch

  supervlad_partial<<<dim3(N_IMG * SPLITS), dim3(256), 0, stream>>>(x, w, ws);
  supervlad_finalize<<<dim3(N_IMG), dim3(256), 0, stream>>>(ws, out);
}
